// DrugEncoder_50972671869198
// MI455X (gfx1250) — matrix operand order repaired and checked by reading, not compiled, not run on hardware
//
#include <hip/hip_runtime.h>
#include <hip/hip_bf16.h>

// ---------------------------------------------------------------------------
// CDNA5 (gfx1250) DrugEncoder (live subgraph only: edge_out is dead code).
// wave32; v_wmma_f32_16x16x32_bf16 for all matmuls (fp32 accumulate);
// GLOBAL_LOAD_ASYNC_TO_LDS_B128 (ASYNCcnt) for weight/tile staging.
// ---------------------------------------------------------------------------

typedef __attribute__((ext_vector_type(16))) __bf16        v16bf;
typedef __attribute__((ext_vector_type(8)))  float         v8f;
typedef __attribute__((ext_vector_type(4)))  unsigned int  u32x4;

union Frag {
    v16bf v;
    u32x4 q[2];
};

__device__ __forceinline__ unsigned short f2bf(float f) {
    union { float f; unsigned u; } c; c.f = f;
    return (unsigned short)(c.u >> 16);   // truncate to bf16
}

// Per-lane async 16B copy: global -> LDS (tracked by ASYNCcnt).
__device__ __forceinline__ void async_copy_b128(void* ldsDst, const void* gSrc) {
    unsigned la = (unsigned)(unsigned long long)ldsDst;     // LDS byte offset (addr[31:0])
    unsigned long long ga = (unsigned long long)gSrc;
    asm volatile("global_load_async_to_lds_b128 %0, %1, off"
                 :: "v"(la), "v"(ga) : "memory");
}
__device__ __forceinline__ void async_wait0() {
    asm volatile("s_wait_asynccnt 0x0" ::: "memory");
}

// A fragment (16x32 bf16, row-major in LDS with bf16 stride 'ld'):
//   elements 0-7 : K = k0 + 8*half + 0..7 ; elements 8-15: K = k0 + 16 + 8*half + 0..7
__device__ __forceinline__ Frag load_fragA(const unsigned short* base, int ld,
                                           int row, int k0, int half) {
    Frag a;
    const unsigned short* p = base + row * ld + k0 + 8 * half;
    a.q[0] = *(const u32x4*)(p);
    a.q[1] = *(const u32x4*)(p + 16);
    return a;
}
// B fragment (32x16 bf16; W stored transposed [n][k], stride 'ld'):
//   elements 0-7 : K = k0 + 8*half + 0..7 ; elements 8-15: K = k0 + 16 + 8*half + 0..7
__device__ __forceinline__ Frag load_fragB(const unsigned short* base, int ld,
                                           int col, int k0, int half) {
    Frag b;
    const unsigned short* p = base + col * ld + k0 + 8 * half;
    b.q[0] = *(const u32x4*)(p);
    b.q[1] = *(const u32x4*)(p + 16);
    return b;
}

// ---------------------------------------------------------------------------
__global__ void zero_ws(float* __restrict__ p, size_t n) {
    size_t i = (size_t)blockIdx.x * blockDim.x + threadIdx.x;
    size_t stride = (size_t)gridDim.x * blockDim.x;
    for (; i < n; i += stride) p[i] = 0.0f;
}

// One-shot: pre-transpose + convert all weights to bf16 in workspace.
__global__ __launch_bounds__(256) void prep_weights(
    const float* __restrict__ W,  const float* __restrict__ W1,
    const float* __restrict__ W2,
    unsigned short* __restrict__ wAT,   // [128][128]  atom_w^T
    unsigned short* __restrict__ w1T,   // [256][128]  nmlp_w1^T
    unsigned short* __restrict__ w2T)   // [128][256]  nmlp_w2^T
{
    int tid = blockIdx.x * blockDim.x + threadIdx.x;
    int stride = gridDim.x * blockDim.x;
    for (int i = tid; i < 128 * 128; i += stride) {
        int n = i >> 7, k = i & 127;
        wAT[n * 128 + k] = f2bf(W[k * 128 + n]);
    }
    for (int i = tid; i < 256 * 128; i += stride) {
        int n = i >> 7, k = i & 127;                    // n<256, k<128
        w1T[n * 128 + k] = f2bf(W1[k * 256 + n]);
    }
    for (int i = tid; i < 128 * 256; i += stride) {
        int n = i >> 8, k = i & 255;                    // n<128, k<256
        w2T[n * 256 + k] = f2bf(W2[k * 128 + n]);
    }
}

// ---------------------------------------------------------------------------
// Y[nrows,128] = X @ W + bias   (W given pre-transposed bf16)
// 64 rows/block, 256 threads (8 waves). Dynamic LDS:
//   sW [128][128] bf16 (32KB) | sA [64][128] bf16 (16KB) |
//   sStage [64][128] f32 (32KB) | sB [128] f32
// ---------------------------------------------------------------------------
__global__ __launch_bounds__(256) void proj_wmma(
    const float* __restrict__ X, const unsigned short* __restrict__ WT,
    const float* __restrict__ bias, float* __restrict__ Y, int nrows)
{
    extern __shared__ char smem[];
    unsigned short* sW = (unsigned short*)smem;
    unsigned short* sA = sW + 128 * 128;
    float* sStage = (float*)(sA + 64 * 128);
    float* sB = sStage + 64 * 128;

    const int tid  = threadIdx.x;
    const int lane = tid & 31;
    const int wave = tid >> 5;
    const int rowBlock = blockIdx.x * 64;
    const bool fullTile = (rowBlock + 64) <= nrows;     // uniform

    // async: W^T bf16 (32KB = 2048 x 16B) straight into LDS
    for (int idx = tid; idx < 2048; idx += 256)
        async_copy_b128(sW + idx * 8, WT + idx * 8);
    // async: A tile fp32 (64 rows x 32 float4) into staging
    for (int idx = tid; idx < 2048; idx += 256) {
        int r = idx >> 5, c = idx & 31;
        int gr = rowBlock + r;
        if (gr < nrows)
            async_copy_b128(sStage + idx * 4, X + (size_t)gr * 128 + c * 4);
    }
    if (tid < 128) sB[tid] = bias[tid];
    async_wait0();
    __syncthreads();

    // convert staged fp32 -> bf16
    for (int i = tid; i < 64 * 128; i += 256) {
        int gr = rowBlock + (i >> 7);
        sA[i] = f2bf(gr < nrows ? sStage[i] : 0.0f);
    }
    __syncthreads();

    const int m = lane & 15, half = lane >> 4;
    // 4 row-tiles x 8 col-tiles = 32 tiles, 4 per wave
    for (int t = wave; t < 32; t += 8) {
        int rowBase = (t >> 3) * 16, colBase = (t & 7) * 16;
        Frag a[4], b[4];
#pragma unroll
        for (int kk = 0; kk < 4; ++kk) {
            a[kk] = load_fragA(sA, 128, rowBase + m, kk * 32, half);
            b[kk] = load_fragB(sW, 128, colBase + m, kk * 32, half);
        }
        v8f acc = {};
#pragma unroll
        for (int kk = 0; kk < 4; ++kk)
            acc = __builtin_amdgcn_wmma_f32_16x16x32_bf16(
                false, a[kk].v, false, b[kk].v, (short)0, acc, false, false);

        int n = colBase + m;
        float bn = sB[n];                          // hoisted bias (one ds load)
        int gm0 = rowBlock + rowBase + 8 * half;
        if (fullTile) {
            float* yp = Y + (size_t)gm0 * 128 + n;
#pragma unroll
            for (int r = 0; r < 8; ++r) yp[(size_t)r * 128] = acc[r] + bn;
        } else {
#pragma unroll
            for (int r = 0; r < 8; ++r) {
                int gm = gm0 + r;
                if (gm < nrows) Y[(size_t)gm * 128 + n] = acc[r] + bn;
            }
        }
    }
}

// ---------------------------------------------------------------------------
// agg[ab_dst[i]] += node_h[ab_src[i]] + edge_h[i]   (one wave per bond)
// ---------------------------------------------------------------------------
__global__ __launch_bounds__(256) void scatter_bonds(
    const float* __restrict__ node_h, const float* __restrict__ edge_h,
    const int* __restrict__ src, const int* __restrict__ dst,
    float* __restrict__ agg, int nbonds)
{
    int bond = blockIdx.x * 8 + (threadIdx.x >> 5);
    if (bond >= nbonds) return;
    int lane = threadIdx.x & 31;
    int s = src[bond], d = dst[bond];
    if (bond + 8 < nbonds)
        __builtin_prefetch(edge_h + (size_t)(bond + 8) * 128 + lane * 4, 0, 0);
    float4 nv = ((const float4*)(node_h + (size_t)s * 128))[lane];
    float4 ev = ((const float4*)(edge_h + (size_t)bond * 128))[lane];
    float* ap = agg + (size_t)d * 128 + lane * 4;
    atomicAdd(ap + 0, nv.x + ev.x);
    atomicAdd(ap + 1, nv.y + ev.y);
    atomicAdd(ap + 2, nv.z + ev.z);
    atomicAdd(ap + 3, nv.w + ev.w);
}

// ---------------------------------------------------------------------------
// Fused: H = relu(agg@W1+b1)@W2+b2 ; h_n = relu(LN(H)) ; out = node_h + h_n ;
//        gsum[gid] += out ; cnt[gid] += 1      (64 rows per block)
// Dynamic LDS (~210 KB):
//   sW1 [256][128] bf16 (64KB) | sW2 [128][256] bf16 (64KB) |
//   sA [64][128] bf16 (16KB) | sT [64][256] bf16 (32KB) |
//   sH [64][128] f32 (32KB, aliased as fp32 staging) | sB1[256] sB2[128]
// ---------------------------------------------------------------------------
__global__ __launch_bounds__(256) void mlp_fused(
    const float* __restrict__ agg, const float* __restrict__ node_h,
    const unsigned short* __restrict__ w1T, const float* __restrict__ B1,
    const unsigned short* __restrict__ w2T, const float* __restrict__ B2,
    const float* __restrict__ G,  const float* __restrict__ Bt,
    const int* __restrict__ gid, float* __restrict__ gsum,
    float* __restrict__ cnt, int nrows)
{
    extern __shared__ char smem[];
    unsigned short* sW1 = (unsigned short*)smem;     // 256x128
    unsigned short* sW2 = sW1 + 256 * 128;           // 128x256
    unsigned short* sA  = sW2 + 128 * 256;           // 64x128
    unsigned short* sT  = sA + 64 * 128;             // 64x256
    float* sH     = (float*)(sT + 64 * 256);         // 64x128 f32
    float* sStage = sH;                              // alias: staging dies before sH written
    float* sB1 = sH + 64 * 128;
    float* sB2 = sB1 + 256;

    const int tid  = threadIdx.x;
    const int lane = tid & 31;
    const int wave = tid >> 5;
    const int rowBlock = blockIdx.x * 64;

    // async: both weight matrices (bf16, pre-transposed) straight into LDS
    for (int idx = tid; idx < 4096; idx += 256)
        async_copy_b128(sW1 + idx * 8, w1T + idx * 8);
    for (int idx = tid; idx < 4096; idx += 256)
        async_copy_b128(sW2 + idx * 8, w2T + idx * 8);
    // async: agg tile fp32 into staging
    for (int idx = tid; idx < 2048; idx += 256) {
        int r = idx >> 5, c = idx & 31;
        int gr = rowBlock + r;
        if (gr < nrows)
            async_copy_b128(sStage + idx * 4, agg + (size_t)gr * 128 + c * 4);
    }
    if (tid < 256) sB1[tid] = B1[tid];
    if (tid < 128) sB2[tid] = B2[tid];
    async_wait0();
    __syncthreads();

    for (int i = tid; i < 64 * 128; i += 256) {
        int gr = rowBlock + (i >> 7);
        sA[i] = f2bf(gr < nrows ? sStage[i] : 0.0f);
    }
    __syncthreads();

    const int m = lane & 15, half = lane >> 4;

    // ---- stage 1: T = relu(A @ W1 + b1), 64x256 : 4x16 tiles, 8 per wave
    for (int t = wave; t < 64; t += 8) {
        int rowBase = (t >> 4) * 16, colBase = (t & 15) * 16;
        Frag a[4], b[4];
#pragma unroll
        for (int kk = 0; kk < 4; ++kk) {
            a[kk] = load_fragA(sA,  128, rowBase + m, kk * 32, half);
            b[kk] = load_fragB(sW1, 128, colBase + m, kk * 32, half);
        }
        v8f acc = {};
#pragma unroll
        for (int kk = 0; kk < 4; ++kk)
            acc = __builtin_amdgcn_wmma_f32_16x16x32_bf16(
                false, a[kk].v, false, b[kk].v, (short)0, acc, false, false);

        int n = colBase + m;
        float b1n = sB1[n];                         // hoisted bias
        int mm0 = rowBase + 8 * half;
#pragma unroll
        for (int r = 0; r < 8; ++r)
            sT[(mm0 + r) * 256 + n] = f2bf(fmaxf(acc[r] + b1n, 0.0f));
    }
    __syncthreads();

    // ---- stage 2: H = T @ W2 + b2, 64x128 : 4x8 tiles, 4 per wave, K=256
    for (int t = wave; t < 32; t += 8) {
        int rowBase = (t >> 3) * 16, colBase = (t & 7) * 16;
        v8f acc = {};
#pragma unroll
        for (int hk = 0; hk < 2; ++hk) {
            Frag a[4], b[4];
#pragma unroll
            for (int kk = 0; kk < 4; ++kk) {
                int k0 = (hk * 4 + kk) * 32;
                a[kk] = load_fragA(sT,  256, rowBase + m, k0, half);
                b[kk] = load_fragB(sW2, 256, colBase + m, k0, half);
            }
#pragma unroll
            for (int kk = 0; kk < 4; ++kk)
                acc = __builtin_amdgcn_wmma_f32_16x16x32_bf16(
                    false, a[kk].v, false, b[kk].v, (short)0, acc, false, false);
        }
        int n = colBase + m;
        float b2n = sB2[n];                         // hoisted bias
        int mm0 = rowBase + 8 * half;
#pragma unroll
        for (int r = 0; r < 8; ++r)
            sH[(mm0 + r) * 128 + n] = acc[r] + b2n;
    }
    __syncthreads();

    // ---- LayerNorm + relu + residual + graph pooling: one wave per row
    for (int r = wave; r < 64; r += 8) {
        int gr = rowBlock + r;
        bool valid = gr < nrows;                // uniform across the wave
        float4 h = make_float4(0.f, 0.f, 0.f, 0.f);
        if (valid) h = ((const float4*)(sH + r * 128))[lane];
        float s = h.x + h.y + h.z + h.w;
        for (int off = 16; off; off >>= 1) s += __shfl_xor(s, off, 32);
        float mu = s * (1.0f / 128.0f);
        float dx = h.x - mu, dy = h.y - mu, dz = h.z - mu, dw = h.w - mu;
        float vv = dx * dx + dy * dy + dz * dz + dw * dw;
        for (int off = 16; off; off >>= 1) vv += __shfl_xor(vv, off, 32);
        float rs = rsqrtf(vv * (1.0f / 128.0f) + 1e-5f);
        if (!valid) continue;
        int g = gid[gr];
        float4 g4 = ((const float4*)G)[lane];
        float4 b4 = ((const float4*)Bt)[lane];
        float4 nh = ((const float4*)(node_h + (size_t)gr * 128))[lane];
        float o0 = nh.x + fmaxf(dx * rs * g4.x + b4.x, 0.0f);
        float o1 = nh.y + fmaxf(dy * rs * g4.y + b4.y, 0.0f);
        float o2 = nh.z + fmaxf(dz * rs * g4.z + b4.z, 0.0f);
        float o3 = nh.w + fmaxf(dw * rs * g4.w + b4.w, 0.0f);
        float* gp = gsum + (size_t)g * 128 + lane * 4;
        atomicAdd(gp + 0, o0);
        atomicAdd(gp + 1, o1);
        atomicAdd(gp + 2, o2);
        atomicAdd(gp + 3, o3);
        if (lane == 0) atomicAdd(&cnt[g], 1.0f);
    }
}

// ---------------------------------------------------------------------------
__global__ void finalize_mean(const float* __restrict__ gsum,
                              const float* __restrict__ cnt,
                              float* __restrict__ out, int total)
{
    int i = blockIdx.x * 256 + threadIdx.x;
    if (i < total) out[i] = gsum[i] / fmaxf(cnt[i >> 7], 1.0f);
}

// ---------------------------------------------------------------------------
extern "C" void kernel_launch(void* const* d_in, const int* in_sizes, int n_in,
                              void* d_out, int out_size, void* d_ws, size_t ws_size,
                              hipStream_t stream)
{
    const float* node_hidden = (const float*)d_in[0];
    const float* edge_hidden = (const float*)d_in[1];
    // d_in[2] angle_hidden: dead (edge_out deleted in reference)
    const float* atom_w = (const float*)d_in[3];
    const float* atom_b = (const float*)d_in[4];
    const float* w1     = (const float*)d_in[5];
    const float* b1     = (const float*)d_in[6];
    const float* w2     = (const float*)d_in[7];
    const float* b2     = (const float*)d_in[8];
    const float* ln_g   = (const float*)d_in[9];
    const float* ln_b   = (const float*)d_in[10];
    const int* ab_src = (const int*)d_in[17];
    const int* ab_dst = (const int*)d_in[18];
    const int* gidx   = (const int*)d_in[21];

    const int H  = 128;
    const int NA = in_sizes[0] / H;   // 500000
    const int NB = in_sizes[1] / H;   // 1000000
    const int NG = out_size / H;      // 16384

    float* node_h = (float*)d_ws;
    float* edge_h = node_h + (size_t)NA * H;
    float* agg    = edge_h + (size_t)NB * H;
    float* gsum   = agg    + (size_t)NA * H;
    float* cnt    = gsum   + (size_t)NG * H;
    unsigned short* wAT = (unsigned short*)(cnt + NG);   // [128][128] bf16
    unsigned short* w1T = wAT + 128 * 128;               // [256][128] bf16
    unsigned short* w2T = w1T + 256 * 128;               // [128][256] bf16

    // zero agg + gsum + cnt (contiguous)
    size_t zn = (size_t)NA * H + (size_t)NG * H + (size_t)NG;
    zero_ws<<<4096, 256, 0, stream>>>(agg, zn);
    prep_weights<<<128, 256, 0, stream>>>(atom_w, w1, w2, wAT, w1T, w2T);

    size_t ldsProj = (size_t)(128 * 128 + 64 * 128) * sizeof(unsigned short)
                   + (size_t)(64 * 128 + 128) * sizeof(float);
    proj_wmma<<<(NA + 63) / 64, 256, ldsProj, stream>>>(node_hidden, wAT, atom_b, node_h, NA);
    proj_wmma<<<(NB + 63) / 64, 256, ldsProj, stream>>>(edge_hidden, wAT, atom_b, edge_h, NB);

    scatter_bonds<<<(NB + 7) / 8, 256, 0, stream>>>(node_h, edge_h, ab_src, ab_dst, agg, NB);

    size_t ldsMlp = (size_t)(256 * 128 + 128 * 256 + 64 * 128 + 64 * 256) * sizeof(unsigned short)
                  + (size_t)(64 * 128 + 256 + 128) * sizeof(float);
    mlp_fused<<<(NA + 63) / 64, 256, ldsMlp, stream>>>(
        agg, node_h, w1T, b1, w2T, b2, ln_g, ln_b, gidx, gsum, cnt, NA);

    finalize_mean<<<(out_size + 255) / 256, 256, 0, stream>>>(gsum, cnt, (float*)d_out, out_size);
}
